// VFunction_25692494365050
// MI455X (gfx1250) — matrix operand order repaired and checked by reading, not compiled, not run on hardware
//
#include <hip/hip_runtime.h>
#include <math.h>

// ---------------- problem constants ----------------
#define BB   1024
#define TT   65
#define DD   512
#define HH   512
#define BT   (BB * TT)            // 66560
#define MROWS 64                  // rows per block
#define LAMF 0.95f
#define DISCF (1.0f - 1.0f / 333.0f)

// ---------------- CDNA5 WMMA vector types ----------------
typedef __attribute__((ext_vector_type(16))) __bf16 v16bf;
typedef __attribute__((ext_vector_type(8)))  __bf16 v8bf;
typedef __attribute__((ext_vector_type(8)))  float  v8f;

__device__ __forceinline__ unsigned short f2bf(float f) {
    unsigned int u = __float_as_uint(f);
    u += 0x7FFFu + ((u >> 16) & 1u);      // round-to-nearest-even
    return (unsigned short)(u >> 16);
}
__device__ __forceinline__ float bf2f(unsigned short h) {
    return __uint_as_float(((unsigned int)h) << 16);
}
__device__ __forceinline__ float silu(float x) {
    return x / (1.0f + __expf(-x));
}

// ---------------------------------------------------------------------------
// Kernel 1: convert + transpose slow-critic weights to bf16 [n][k] so WMMA
// B-fragments (dense 32x16 bf16: lane n = L&15, this code's K order (same as A): 16 contiguous K per half-wave)
// are single contiguous 32B loads.
// ---------------------------------------------------------------------------
__global__ void __launch_bounds__(256)
prep_weights(const float* __restrict__ W1, const float* __restrict__ W2,
             unsigned short* __restrict__ W1t, unsigned short* __restrict__ W2t) {
    int idx = blockIdx.x * 256 + threadIdx.x;   // 0 .. 512*512-1
    int n = idx >> 9;
    int k = idx & 511;
    W1t[idx] = f2bf(W1[k * HH + n]);            // W is [k][n] row-major
    W2t[idx] = f2bf(W2[k * HH + n]);
}

// ---------------------------------------------------------------------------
// Kernel 2: fused 3-layer MLP (slow critic) + vscale/voffset -> tarval.
// Block = 256 threads = 8 wave32, owns 64 rows.
// Waves are DISJOINT in N (each owns a 64-col strip = 4 N-tiles) and loop
// over all 4 M-tiles, so every weight byte is fetched from L2 exactly once
// per block (4x less L2 traffic than an M-split). Per K-step: 4 B (global)
// + 4 A (LDS) fragment loads feed 16 back-to-back
// v_wmma_f32_16x16x32_bf16, with both operands double-buffered one K-step
// ahead. LDS: two 64x512 bf16 buffers (128 KB), ping-ponged between layers.
// ---------------------------------------------------------------------------
__global__ void __launch_bounds__(256)
mlp64(const float* __restrict__ feat,
      const unsigned short* __restrict__ W1t, const float* __restrict__ b1,
      const unsigned short* __restrict__ W2t, const float* __restrict__ b2,
      const float* __restrict__ W3,  const float* __restrict__ b3,
      const float* __restrict__ vscale, const float* __restrict__ voffset,
      float* __restrict__ tarval)
{
    __shared__ unsigned short sA[MROWS * 512];   // L1 input, L2 output
    __shared__ unsigned short sH[MROWS * 512];   // L1 output, L2 input

    const int tid   = threadIdx.x;
    const int lane  = tid & 31;
    const int w     = tid >> 5;           // wave 0..7
    const int nl    = lane & 15;
    const int hh    = lane >> 4;          // half-wave select
    const int rbase = blockIdx.x * MROWS;
    const int nb    = w * 64;             // wave's private 64-col N-strip
    // A-frag base (bf16 index) for M-tile m: (m*16 + nl)*512 + 16*hh
    const int abase = nl * 512 + 16 * hh;

    // ---- stage feat tile (f32 -> bf16) into sA, row-major [m][k] ----
    const float4* fp = reinterpret_cast<const float4*>(feat + (size_t)rbase * DD);
    #pragma unroll
    for (int it = 0; it < 32; ++it) {
        int e4 = it * 256 + tid;                        // 0..8191 float4's
        float4 v = fp[e4];
        uint2 pk;
        pk.x = (unsigned)f2bf(v.x) | ((unsigned)f2bf(v.y) << 16);
        pk.y = (unsigned)f2bf(v.z) | ((unsigned)f2bf(v.w) << 16);
        *reinterpret_cast<uint2*>(&sA[e4 * 4]) = pk;
    }
    __syncthreads();

    // =================== layer 1: sA @ W1t -> silu -> sH ===================
    {
        const unsigned short* wb = W1t + (size_t)(nb + nl) * 512 + 16 * hh;
        v8f acc[4][4] = {};                      // [m-tile][n-tile]
        v16bf afrag[2][4], bfrag[2][4];
        // prologue: loads for k = 0
        #pragma unroll
        for (int m = 0; m < 4; ++m) {
            v8bf lo = *reinterpret_cast<const v8bf*>(&sA[m * 8192 + abase]);
            v8bf hi = *reinterpret_cast<const v8bf*>(&sA[m * 8192 + abase + 8]);
            afrag[0][m] = __builtin_shufflevector(lo, hi,
                            0,1,2,3,4,5,6,7,8,9,10,11,12,13,14,15);
        }
        #pragma unroll
        for (int t = 0; t < 4; ++t)
            bfrag[0][t] = *reinterpret_cast<const v16bf*>(&wb[(size_t)t * 8192]);
        #pragma unroll
        for (int k = 0; k < 16; ++k) {
            const int cur = k & 1, nxt = cur ^ 1;
            if (k < 15) {                        // issue next K-step's loads
                const int k1 = (k + 1) * 32;
                #pragma unroll
                for (int m = 0; m < 4; ++m) {
                    v8bf lo = *reinterpret_cast<const v8bf*>(
                                  &sA[m * 8192 + abase + k1]);
                    v8bf hi = *reinterpret_cast<const v8bf*>(
                                  &sA[m * 8192 + abase + k1 + 8]);
                    afrag[nxt][m] = __builtin_shufflevector(lo, hi,
                                      0,1,2,3,4,5,6,7,8,9,10,11,12,13,14,15);
                }
                #pragma unroll
                for (int t = 0; t < 4; ++t)
                    bfrag[nxt][t] = *reinterpret_cast<const v16bf*>(
                                        &wb[(size_t)t * 8192 + k1]);
            }
            #pragma unroll
            for (int m = 0; m < 4; ++m)
                #pragma unroll
                for (int t = 0; t < 4; ++t)
                    acc[m][t] = __builtin_amdgcn_wmma_f32_16x16x32_bf16(
                                    false, afrag[cur][m], false, bfrag[cur][t],
                                    (short)0, acc[m][t], false, false);
        }
        #pragma unroll
        for (int m = 0; m < 4; ++m)
            #pragma unroll
            for (int t = 0; t < 4; ++t) {
                int n = nb + t * 16 + nl;
                float bias = b1[n];
                #pragma unroll
                for (int r = 0; r < 8; ++r) {    // C: row = m*16 + 8*hh + r
                    float x = acc[m][t][r] + bias;
                    sH[(m * 16 + 8 * hh + r) * 512 + n] = f2bf(silu(x));
                }
            }
    }
    __syncthreads();

    // =================== layer 2: sH @ W2t -> silu -> sA ===================
    {
        const unsigned short* wb = W2t + (size_t)(nb + nl) * 512 + 16 * hh;
        v8f acc[4][4] = {};
        v16bf afrag[2][4], bfrag[2][4];
        #pragma unroll
        for (int m = 0; m < 4; ++m) {
            v8bf lo = *reinterpret_cast<const v8bf*>(&sH[m * 8192 + abase]);
            v8bf hi = *reinterpret_cast<const v8bf*>(&sH[m * 8192 + abase + 8]);
            afrag[0][m] = __builtin_shufflevector(lo, hi,
                            0,1,2,3,4,5,6,7,8,9,10,11,12,13,14,15);
        }
        #pragma unroll
        for (int t = 0; t < 4; ++t)
            bfrag[0][t] = *reinterpret_cast<const v16bf*>(&wb[(size_t)t * 8192]);
        #pragma unroll
        for (int k = 0; k < 16; ++k) {
            const int cur = k & 1, nxt = cur ^ 1;
            if (k < 15) {
                const int k1 = (k + 1) * 32;
                #pragma unroll
                for (int m = 0; m < 4; ++m) {
                    v8bf lo = *reinterpret_cast<const v8bf*>(
                                  &sH[m * 8192 + abase + k1]);
                    v8bf hi = *reinterpret_cast<const v8bf*>(
                                  &sH[m * 8192 + abase + k1 + 8]);
                    afrag[nxt][m] = __builtin_shufflevector(lo, hi,
                                      0,1,2,3,4,5,6,7,8,9,10,11,12,13,14,15);
                }
                #pragma unroll
                for (int t = 0; t < 4; ++t)
                    bfrag[nxt][t] = *reinterpret_cast<const v16bf*>(
                                        &wb[(size_t)t * 8192 + k1]);
            }
            #pragma unroll
            for (int m = 0; m < 4; ++m)
                #pragma unroll
                for (int t = 0; t < 4; ++t)
                    acc[m][t] = __builtin_amdgcn_wmma_f32_16x16x32_bf16(
                                    false, afrag[cur][m], false, bfrag[cur][t],
                                    (short)0, acc[m][t], false, false);
        }
        #pragma unroll
        for (int m = 0; m < 4; ++m)
            #pragma unroll
            for (int t = 0; t < 4; ++t) {
                int n = nb + t * 16 + nl;
                float bias = b2[n];
                #pragma unroll
                for (int r = 0; r < 8; ++r) {
                    float x = acc[m][t][r] + bias;
                    sA[(m * 16 + 8 * hh + r) * 512 + n] = f2bf(silu(x));
                }
            }
    }
    __syncthreads();

    // ===== layer 3: 4 lanes per row dot with W3, shuffle-reduce, scale =====
    {
        int r = tid >> 2;            // 0..63
        int q = tid & 3;             // quarter of the 512-dot
        float sum = 0.0f;
        const int k0 = q * 128;
        #pragma unroll 8
        for (int k = 0; k < 128; ++k)
            sum = fmaf(bf2f(sA[r * 512 + k0 + k]), W3[k0 + k], sum);
        sum += __shfl_xor(sum, 1);
        sum += __shfl_xor(sum, 2);
        if (q == 0) {
            float out = sum + b3[0];
            tarval[rbase + r] = out * vscale[0] + voffset[0];
        }
    }
}

// ---------------------------------------------------------------------------
// Kernel 3: reverse lambda-return scan, one thread per batch row.
// ---------------------------------------------------------------------------
__global__ void __launch_bounds__(256)
lam_scan(const float* __restrict__ tarval, const float* __restrict__ reward,
         const float* __restrict__ cont, float* __restrict__ out)
{
    int b = blockIdx.x * 256 + threadIdx.x;
    if (b >= BB) return;
    const float* tv = tarval + (size_t)b * TT;
    const float* rw = reward + (size_t)b * TT;
    const float* cn = cont   + (size_t)b * TT;
    float carry = tv[TT - 1];
    for (int t = TT - 2; t >= 0; --t) {
        float d = cn[t + 1] * DISCF;
        float i = rw[t + 1] + (1.0f - LAMF) * d * tv[t + 1];
        carry = i + d * LAMF * carry;
        out[(size_t)b * (TT - 1) + t] = carry;
    }
}

// ---------------------------------------------------------------------------
extern "C" void kernel_launch(void* const* d_in, const int* in_sizes, int n_in,
                              void* d_out, int out_size, void* d_ws, size_t ws_size,
                              hipStream_t stream)
{
    const float* feat    = (const float*)d_in[0];
    const float* reward  = (const float*)d_in[1];
    const float* cont    = (const float*)d_in[2];
    const float* voffset = (const float*)d_in[3];
    const float* vscale  = (const float*)d_in[4];
    // fast-critic weights (d_in[5..10]) are dead code in the reference output
    const float* W1s = (const float*)d_in[11];
    const float* b1s = (const float*)d_in[12];
    const float* W2s = (const float*)d_in[13];
    const float* b2s = (const float*)d_in[14];
    const float* W3s = (const float*)d_in[15];
    const float* b3s = (const float*)d_in[16];

    unsigned short* W1t = (unsigned short*)d_ws;            // 512 KB
    unsigned short* W2t = W1t + DD * HH;                    // 512 KB
    float*          tv  = (float*)(W2t + HH * HH);          // 260 KB tarval

    prep_weights<<<(DD * HH) / 256, 256, 0, stream>>>(W1s, W2s, W1t, W2t);
    mlp64<<<BT / MROWS, 256, 0, stream>>>(feat, W1t, b1s, W2t, b2s, W3s, b3s,
                                          vscale, voffset, tv);
    lam_scan<<<(BB + 255) / 256, 256, 0, stream>>>(tv, reward, cont, (float*)d_out);
}
